// GCN_15590731285057
// MI455X (gfx1250) — matrix operand order repaired and checked by reading, not compiled, not run on hardware
//
#include <hip/hip_runtime.h>

#define DF    128      // feature dim
#define LDSTR 136      // padded LDS row stride in halves (272B, kills 256B bank aliasing)
#define FOSTR 132      // padded f32 epilogue-tile stride in floats

typedef __attribute__((ext_vector_type(16))) _Float16 v16h;
typedef __attribute__((ext_vector_type(8)))  _Float16 v8h;
typedef __attribute__((ext_vector_type(4)))  _Float16 v4h;
typedef __attribute__((ext_vector_type(8)))  float    v8f;

static __device__ __forceinline__ v16h cat16(v8h lo, v8h hi) {
  v16h r;
#pragma unroll
  for (int i = 0; i < 8; ++i) { r[i] = lo[i]; r[i + 8] = hi[i]; }
  return r;
}

// ---------------- utility kernels ----------------

__global__ void zero_f32(float* __restrict__ p, long n) {
  long i = (long)blockIdx.x * blockDim.x + threadIdx.x;
  if (i < n) p[i] = 0.0f;
}

__global__ void zero_f32_v4(float4* __restrict__ p, long n4) {
  long i = (long)blockIdx.x * blockDim.x + threadIdx.x;
  if (i < n4) p[i] = make_float4(0.f, 0.f, 0.f, 0.f);
}

__global__ void degrees_kernel(const int* __restrict__ src, const int* __restrict__ dst,
                               float* __restrict__ degO, float* __restrict__ degI, int E) {
  int e = blockIdx.x * blockDim.x + threadIdx.x;
  if (e < E) {
    atomicAdd(&degO[src[e]], 1.0f);
    atomicAdd(&degI[dst[e]], 1.0f);
  }
}

__global__ void rsqrt_kernel(float* __restrict__ p, int n) {
  int i = blockIdx.x * blockDim.x + threadIdx.x;
  if (i < n) p[i] = rsqrtf(fmaxf(p[i], 1.0f));
}

// W (f32, row-major [DF,DF]) -> f16, once per layer
__global__ void w_to_half(const float* __restrict__ W, _Float16* __restrict__ Wh) {
  int i = blockIdx.x * blockDim.x + threadIdx.x;
  if (i < DF * DF) Wh[i] = (_Float16)W[i];
}

// ---------------- SpMM: agg[dst] += x[src] * norm_out[src] ----------------
// One edge per wave32: 32 lanes x float4 = 128 floats (512B coalesced gather),
// scatter via f32 global atomics (L2-resident accumulator).
__global__ void __launch_bounds__(256) spmm_kernel(const float* __restrict__ x,
                                                   const float* __restrict__ normO,
                                                   const int* __restrict__ src,
                                                   const int* __restrict__ dst,
                                                   float* __restrict__ agg, int E) {
  int e = blockIdx.x * 8 + (threadIdx.x >> 5);
  if (e >= E) return;
  int lane = threadIdx.x & 31;
  int s = src[e];
  int d = dst[e];
  float ns = normO[s];
  float4 v = *((const float4*)(x + (long)s * DF) + lane);
  float* ar = agg + (long)d * DF + lane * 4;
  atomicAdd(ar + 0, v.x * ns);
  atomicAdd(ar + 1, v.y * ns);
  atomicAdd(ar + 2, v.z * ns);
  atomicAdd(ar + 3, v.w * ns);
}

// ---------------- GEMM + epilogue (WMMA f16 -> f32) ----------------
// out[r,:] = act( (agg[r,:] @ W^T + b) * norm_in[r] (+ resid[r,:]) )
// Block: 128 threads = 4 waves; block tile = 64 rows x 128 cols.
// Each wave owns 16 rows; column tiles processed in pairs (2 indep wmma chains).
// Epilogue: C-fragments -> LDS f32 tile -> coalesced float4 global traffic.
template <bool RESID, bool RELU>
__global__ void __launch_bounds__(128) gcn_gemm_wmma(const float* __restrict__ agg,
                                                     const _Float16* __restrict__ Wh,
                                                     const float* __restrict__ bias,
                                                     const float* __restrict__ normI,
                                                     const float* __restrict__ resid,
                                                     float* __restrict__ out,
                                                     int n) {
  // [ sW: 128x136 halves | sA: 64x136 halves ]; f32 epilogue tile overlays sA.
  __shared__ __align__(16) _Float16 smemh[DF * LDSTR + 64 * LDSTR];
  _Float16* sW = smemh;
  _Float16* sA = smemh + DF * LDSTR;
  float*    fout = (float*)sA;                 // 64 x FOSTR floats (33.0 KB; sA holds only 17 KB)

  const int tid = threadIdx.x;
  const int rows0 = blockIdx.x * 64;

  // Stage W (already f16): 16B vector copies, coalesced.
#pragma unroll
  for (int it = 0; it < 16; ++it) {
    int ci = tid + it * 128;                   // 16-byte chunk index
    int base = ci * 8;
    int r = base >> 7, c = base & 127;
    *(v8h*)&sW[r * LDSTR + c] = *(const v8h*)&Wh[base];
  }
  // Stage A tile: float4 load (coalesced) -> 4 halves, zero-pad past n.
#pragma unroll
  for (int it = 0; it < 16; ++it) {
    int q = tid + it * 128;                    // float4 index
    int base = q * 4;
    int r = base >> 7, c = base & 127;
    int gr = rows0 + r;
    float4 v = (gr < n) ? *(const float4*)&agg[(long)gr * DF + c]
                        : make_float4(0.f, 0.f, 0.f, 0.f);
    v4h h; h[0] = (_Float16)v.x; h[1] = (_Float16)v.y;
           h[2] = (_Float16)v.z; h[3] = (_Float16)v.w;
    *(v4h*)&sA[r * LDSTR + c] = h;
  }
  __syncthreads();

  const int lane  = tid & 31;   // wave32
  const int wv    = tid >> 5;   // wave id 0..3
  const int lhalf = lane & 15;
  const int lhi   = lane >> 4;
  const int arow  = wv * 16 + lhalf;

  // A fragments for all 4 K-steps (contiguous k order, same as B):
  // lane L: row = L&15; K contiguous 16*(L>>4)+{0..15}
  v16h afrag[4];
#pragma unroll
  for (int ks = 0; ks < 4; ++ks) {
    const _Float16* pa = &sA[arow * LDSTR + ks * 32 + lhi * 16];
    v8h lo = *(const v8h*)pa;          // 16B aligned (stride 272B)
    v8h hi = *(const v8h*)(pa + 8);
    afrag[ks] = cat16(lo, hi);
  }
  __syncthreads();                     // sA dead; its space becomes fout

  // Column tiles in pairs: two independent accumulator chains overlap wmma issue.
#pragma unroll
  for (int jp = 0; jp < 4; ++jp) {
    const int jt0 = jp * 2, jt1 = jp * 2 + 1;
    v8f acc0 = {}, acc1 = {};
#pragma unroll
    for (int ks = 0; ks < 4; ++ks) {
      // B layout: lane L holds col N=L&15, K contiguous 16*(L>>4)+{0..15};
      // B[k][n] = W[n][k], W row-major -> contiguous along k.
      const _Float16* pb0 = &sW[(jt0 * 16 + lhalf) * LDSTR + ks * 32 + lhi * 16];
      const _Float16* pb1 = &sW[(jt1 * 16 + lhalf) * LDSTR + ks * 32 + lhi * 16];
      v16h b0 = cat16(*(const v8h*)pb0, *(const v8h*)(pb0 + 8));
      v16h b1 = cat16(*(const v8h*)pb1, *(const v8h*)(pb1 + 8));
      acc0 = __builtin_amdgcn_wmma_f32_16x16x32_f16(false, afrag[ks], false, b0,
                                                    (short)0, acc0, false, false);
      acc1 = __builtin_amdgcn_wmma_f32_16x16x32_f16(false, afrag[ks], false, b1,
                                                    (short)0, acc1, false, false);
    }
    // C/D layout: VGPR g -> row g + 8*(lane>>4); col = lane&15 (+16*jt)
    const int rb = wv * 16 + lhi * 8;
#pragma unroll
    for (int g = 0; g < 8; ++g) {
      fout[(rb + g) * FOSTR + jt0 * 16 + lhalf] = acc0[g];
      fout[(rb + g) * FOSTR + jt1 * 16 + lhalf] = acc1[g];
    }
  }
  __syncthreads();

  // Coalesced epilogue: float4 loads/stores, compile-time RESID/RELU.
#pragma unroll
  for (int it = 0; it < 16; ++it) {
    int q = tid + it * 128;                    // float4 index within 64x128 tile
    int row = q >> 5;                          // 32 float4 per row
    int col = (q & 31) * 4;
    int gr = rows0 + row;
    if (gr < n) {
      float4 a  = *(const float4*)&fout[row * FOSTR + col];
      float4 bb = *(const float4*)&bias[col];
      float ni = normI[gr];
      float4 v;
      v.x = (a.x + bb.x) * ni;
      v.y = (a.y + bb.y) * ni;
      v.z = (a.z + bb.z) * ni;
      v.w = (a.w + bb.w) * ni;
      if (RESID) {
        float4 rr = *(const float4*)&resid[(long)gr * DF + col];
        v.x += rr.x; v.y += rr.y; v.z += rr.z; v.w += rr.w;
      }
      if (RELU) {
        v.x = fmaxf(v.x, 0.f); v.y = fmaxf(v.y, 0.f);
        v.z = fmaxf(v.z, 0.f); v.w = fmaxf(v.w, 0.f);
      }
      *(float4*)&out[(long)gr * DF + col] = v;
    }
  }
}

// ---------------- host launcher ----------------
extern "C" void kernel_launch(void* const* d_in, const int* in_sizes, int n_in,
                              void* d_out, int out_size, void* d_ws, size_t ws_size,
                              hipStream_t stream) {
  (void)n_in; (void)out_size; (void)ws_size;
  const float* feat = (const float*)d_in[0];
  const int*   src  = (const int*)d_in[1];
  const int*   dst  = (const int*)d_in[2];
  const float* W1 = (const float*)d_in[3];
  const float* b1 = (const float*)d_in[4];
  const float* W2 = (const float*)d_in[5];
  const float* b2 = (const float*)d_in[6];
  const float* W3 = (const float*)d_in[7];
  const float* b3 = (const float*)d_in[8];

  const int N = in_sizes[0] / DF;
  const int E = in_sizes[1];
  float* out = (float*)d_out;

  // workspace: [normO N][normI N][agg N*DF][xA N*DF][Wh DF*DF halves]
  float* normO = (float*)d_ws;
  float* normI = normO + N;
  float* agg   = normI + N;
  float* xA    = agg + (size_t)N * DF;
  _Float16* Wh = (_Float16*)(xA + (size_t)N * DF);

  const long nd4   = (long)N * DF / 4;
  const int  gZ4   = (int)((nd4 + 255) / 256);
  const int  gSpmm = (E + 7) / 8;
  const int  gGemm = (N + 63) / 64;
  const int  gW    = (DF * DF + 255) / 256;

  // degree norms (shared by all layers)
  zero_f32<<<(2 * N + 255) / 256, 256, 0, stream>>>(normO, 2L * N);
  degrees_kernel<<<(E + 255) / 256, 256, 0, stream>>>(src, dst, normO, normI, E);
  rsqrt_kernel<<<(2 * N + 255) / 256, 256, 0, stream>>>(normO, 2 * N);

  // layer 1: feat -> d_out  (residual=feat, relu)
  zero_f32_v4<<<gZ4, 256, 0, stream>>>((float4*)agg, nd4);
  w_to_half<<<gW, 256, 0, stream>>>(W1, Wh);
  spmm_kernel<<<gSpmm, 256, 0, stream>>>(feat, normO, src, dst, agg, E);
  gcn_gemm_wmma<true, true><<<gGemm, 128, 0, stream>>>(agg, Wh, b1, normI, feat, out, N);

  // layer 2: d_out -> xA  (residual=d_out, relu)
  zero_f32_v4<<<gZ4, 256, 0, stream>>>((float4*)agg, nd4);
  w_to_half<<<gW, 256, 0, stream>>>(W2, Wh);
  spmm_kernel<<<gSpmm, 256, 0, stream>>>(out, normO, src, dst, agg, E);
  gcn_gemm_wmma<true, true><<<gGemm, 128, 0, stream>>>(agg, Wh, b2, normI, out, xA, N);

  // layer 3: xA -> d_out  (no residual, no act)
  zero_f32_v4<<<gZ4, 256, 0, stream>>>((float4*)agg, nd4);
  w_to_half<<<gW, 256, 0, stream>>>(W3, Wh);
  spmm_kernel<<<gSpmm, 256, 0, stream>>>(xA, normO, src, dst, agg, E);
  gcn_gemm_wmma<false, false><<<gGemm, 128, 0, stream>>>(agg, Wh, b3, normI, nullptr, out, N);
}
